// AttentionModule_20822001451476
// MI455X (gfx1250) — matrix operand order repaired and checked by reading, not compiled, not run on hardware
//
#include <hip/hip_runtime.h>

typedef __attribute__((ext_vector_type(8)))  __bf16 v8bf;
typedef __attribute__((ext_vector_type(16))) __bf16 v16bf;
typedef __attribute__((ext_vector_type(8)))  float  v8f;

#define D_MODEL 1024
#define N_TOK   2048
#define N_HEAD  16
#define HEAD_D  64

// Padded LDS row pitches (in bf16 elements): 40 elems = 80 B = 5*16 B,
// 72 elems = 144 B = 9*16 B -> both 16B-aligned chunks and odd-16B bank stride.
#define LDA 40
#define LDB 40
#define LDK 72
#define LDV 40

__device__ __forceinline__ __bf16 f2bf(float f) { return (__bf16)f; }

__device__ __forceinline__ v8f wmma_bf16(v16bf a, v16bf b, v8f c) {
  // D = A(16x32 bf16) * B(32x16 bf16) + C(16x16 f32)
  return __builtin_amdgcn_wmma_f32_16x16x32_bf16(false, a, false, b, (short)0, c,
                                                 false, false);
}

// Async DMA: 16 bytes per lane, global -> LDS, tracked by ASYNCcnt.
__device__ __forceinline__ void async_copy16(void* lds_ptr, const void* gptr) {
  unsigned loff = (unsigned)(size_t)lds_ptr;       // low 32 bits = LDS offset
  asm volatile("global_load_async_to_lds_b128 %0, %1, off"
               :: "v"(loff), "v"(gptr)
               : "memory");
}
__device__ __forceinline__ void wait_async0() {
  asm volatile("s_wait_asynccnt 0x0" ::: "memory");
}

// A-fragment 16x32 bf16 (MxK). Lanes 0-15 rows M=lane&15, K contiguous
// {g*16 .. g*16+15}, g = lane>>4.
__device__ __forceinline__ v16bf load_a(const __bf16* p, int row0, int ld, int k0, int lane) {
  int r = row0 + (lane & 15);
  int g = (lane >> 4) * 16;
  const __bf16* rp = p + (size_t)r * ld + k0 + g;
  v8bf lo = *(const v8bf*)rp;
  v8bf hi = *(const v8bf*)(rp + 8);
  v16bf out;
#pragma unroll
  for (int i = 0; i < 8; ++i) { out[i] = lo[i]; out[i + 8] = hi[i]; }
  return out;
}

// B-fragment 32x16 bf16 (KxN). Lane holds column col0+(lane&15);
// lanes 0-15: K=k0..k0+15, lanes 16-31: K=k0+16..k0+31.
__device__ __forceinline__ v16bf load_b(const __bf16* p, int col0, int ld, int k0, int lane) {
  int c = col0 + (lane & 15);
  int kk = k0 + ((lane >> 4) << 4);
  return *(const v16bf*)(p + (size_t)c * ld + kk);
}

// ---------------------------------------------------------------- conversion
__global__ void cvt_f32_to_bf16(const float* __restrict__ in, __bf16* __restrict__ out, int n) {
  int i = blockIdx.x * blockDim.x + threadIdx.x;
  int stride = gridDim.x * blockDim.x;
  for (; i < n; i += stride) out[i] = f2bf(in[i]);
}

// ----------------------------------------------------- fused QKV projections
// Block tile 64(M) x 128(N); 8 waves of 32x32. A/B k-slabs double-buffered in
// LDS via async global->LDS DMA. grid.z: 0->Q, 1->K, 2->V (V transposed out).
__global__ void qkv_gemm(const __bf16* __restrict__ xb,
                         const __bf16* __restrict__ wq, const __bf16* __restrict__ wk,
                         const __bf16* __restrict__ wv,
                         const float* __restrict__ bq, const float* __restrict__ bk,
                         const float* __restrict__ bv,
                         __bf16* __restrict__ qo, __bf16* __restrict__ ko,
                         __bf16* __restrict__ vTo) {
  __shared__ __align__(16) __bf16 sA[2][64 * LDA];    // 64 rows x 32 K
  __shared__ __align__(16) __bf16 sB[2][128 * LDB];   // 128 cols x 32 K
  const int z = blockIdx.z;
  const __bf16* X   = xb;
  const __bf16* W   = (z == 0) ? wq : (z == 1) ? wk : wv;
  const float* bias = (z == 0) ? bq : (z == 1) ? bk : bv;
  const int tid  = threadIdx.x;
  const int lane = tid & 31;
  const int warp = tid >> 5;
  const int m0blk = (blockIdx.x >> 3) * 64;    // 32 M-blocks
  const int n0blk = (blockIdx.x & 7) * 128;    // 8 N-blocks
  const int mw = warp >> 2;                    // 0..1
  const int nw = warp & 3;                     // 0..3
  const int m0 = m0blk + mw * 32;
  const int n0 = n0blk + nw * 32;
  const int row4 = tid >> 2;                   // 0..63
  const int ch   = tid & 3;                    // 16B chunk within 32 K elems

  auto stage = [&](int buf, int k0) {
    async_copy16(&sA[buf][row4 * LDA + ch * 8],
                 X + (size_t)(m0blk + row4) * D_MODEL + k0 + ch * 8);
    async_copy16(&sB[buf][row4 * LDB + ch * 8],
                 W + (size_t)(n0blk + row4) * D_MODEL + k0 + ch * 8);
    async_copy16(&sB[buf][(64 + row4) * LDB + ch * 8],
                 W + (size_t)(n0blk + 64 + row4) * D_MODEL + k0 + ch * 8);
  };

  v8f c00 = {}, c01 = {}, c10 = {}, c11 = {};
  int buf = 0;
  stage(0, 0);
  for (int k0 = 0; k0 < D_MODEL; k0 += 32) {
    wait_async0();
    __syncthreads();
    if (k0 + 32 < D_MODEL) stage(buf ^ 1, k0 + 32);
    v16bf a0 = load_a(&sA[buf][0], mw * 32,      LDA, 0, lane);
    v16bf a1 = load_a(&sA[buf][0], mw * 32 + 16, LDA, 0, lane);
    v16bf b0 = load_b(&sB[buf][0], nw * 32,      LDB, 0, lane);
    v16bf b1 = load_b(&sB[buf][0], nw * 32 + 16, LDB, 0, lane);
    c00 = wmma_bf16(a0, b0, c00);
    c01 = wmma_bf16(a0, b1, c01);
    c10 = wmma_bf16(a1, b0, c10);
    c11 = wmma_bf16(a1, b1, c11);
    __syncthreads();
    buf ^= 1;
  }

  const int colb = lane & 15;
  const int rowoff = (lane >> 4) * 8;
  v8f cf[2][2] = {{c00, c01}, {c10, c11}};
  if (z < 2) {
    __bf16* out = z ? ko : qo;
#pragma unroll
    for (int mi = 0; mi < 2; ++mi)
#pragma unroll
      for (int ni = 0; ni < 2; ++ni)
#pragma unroll
        for (int i = 0; i < 8; ++i) {
          int row = m0 + mi * 16 + rowoff + i;
          int col = n0 + ni * 16 + colb;
          out[(size_t)row * D_MODEL + col] = f2bf(cf[mi][ni][i] + bias[col]);
        }
  } else {
#pragma unroll
    for (int mi = 0; mi < 2; ++mi)
#pragma unroll
      for (int ni = 0; ni < 2; ++ni)
#pragma unroll
        for (int i = 0; i < 8; ++i) {
          int row = m0 + mi * 16 + rowoff + i;
          int col = n0 + ni * 16 + colb;
          vTo[(size_t)col * N_TOK + row] = f2bf(cf[mi][ni][i] + bias[col]);
        }
  }
}

// ------------------------------------------------------------ flash attention
// Block = 8 waves on the SAME head, 128 query rows (16/wave). K/V tiles for
// each 32-key step are async-staged into LDS once and shared by all 8 waves.
__global__ void attn_kernel(const __bf16* __restrict__ qb, const __bf16* __restrict__ kb,
                            const __bf16* __restrict__ vT, __bf16* __restrict__ ob) {
  __shared__ __align__(16) __bf16 sK[2][32 * LDK];   // 32 keys x 64 dims
  __shared__ __align__(16) __bf16 sV[2][64 * LDV];   // 64 dims x 32 keys
  __shared__ __align__(16) __bf16 plds[8 * 16 * 32]; // per-wave P tile
  const int tid  = threadIdx.x;
  const int lane = tid & 31;
  const int warp = tid >> 5;
  const int h    = blockIdx.x >> 4;            // 16 q-blocks per head
  const int m0   = (blockIdx.x & 15) * 128 + warp * 16;
  __bf16* pbuf = plds + warp * (16 * 32);

  auto stage = [&](int buf, int j0) {
    // K rows j0..j0+31, head slice (64 dims): thread t -> row t>>3, chunk t&7
    async_copy16(&sK[buf][(tid >> 3) * LDK + (tid & 7) * 8],
                 kb + (size_t)(j0 + (tid >> 3)) * D_MODEL + h * HEAD_D + (tid & 7) * 8);
    // V^T rows h*64..h*64+63, keys j0..j0+31: thread t -> row t>>2, chunk t&3
    async_copy16(&sV[buf][(tid >> 2) * LDV + (tid & 3) * 8],
                 vT + (size_t)(h * HEAD_D + (tid >> 2)) * N_TOK + j0 + (tid & 3) * 8);
  };

  // Preload Q A-fragments (16 rows x 64 head-dims)
  v16bf qa0 = load_a(qb, m0, D_MODEL, h * HEAD_D,      lane);
  v16bf qa1 = load_a(qb, m0, D_MODEL, h * HEAD_D + 32, lane);

  v8f o0 = {}, o1 = {}, o2 = {}, o3 = {};
  float mrun[8], lrun[8];
#pragma unroll
  for (int i = 0; i < 8; ++i) { mrun[i] = -1.0e30f; lrun[i] = 0.0f; }
  const float scale = 1.0f / (float)HEAD_D;    // reference divides by hd
  const int colb = lane & 15;
  const int rowoff = (lane >> 4) * 8;

  int buf = 0;
  stage(0, 0);
  for (int j0 = 0; j0 < N_TOK; j0 += 32) {
    wait_async0();
    __syncthreads();
    if (j0 + 32 < N_TOK) stage(buf ^ 1, j0 + 32);

    // scores S(16x32) = Q(16x64) x K^T(64x32) from LDS K tile
    v8f s0 = {}, s1 = {};
    {
      v16bf kb00 = load_b(&sK[buf][0], 0,  LDK, 0,  lane);
      v16bf kb01 = load_b(&sK[buf][0], 0,  LDK, 32, lane);
      s0 = wmma_bf16(qa0, kb00, s0);
      s0 = wmma_bf16(qa1, kb01, s0);
      v16bf kb10 = load_b(&sK[buf][0], 16, LDK, 0,  lane);
      v16bf kb11 = load_b(&sK[buf][0], 16, LDK, 32, lane);
      s1 = wmma_bf16(qa0, kb10, s1);
      s1 = wmma_bf16(qa1, kb11, s1);
    }
    float corr[8];
#pragma unroll
    for (int i = 0; i < 8; ++i) {
      float a = s0[i] * scale;
      float b = s1[i] * scale;
      float tm = fmaxf(a, b);
#pragma unroll
      for (int off = 1; off < 16; off <<= 1) tm = fmaxf(tm, __shfl_xor(tm, off, 32));
      float mn = fmaxf(mrun[i], tm);
      corr[i] = __expf(mrun[i] - mn);
      mrun[i] = mn;
      float p0 = __expf(a - mn);
      float p1 = __expf(b - mn);
      s0[i] = p0; s1[i] = p1;
      float rs = p0 + p1;
#pragma unroll
      for (int off = 1; off < 16; off <<= 1) rs += __shfl_xor(rs, off, 32);
      lrun[i] = lrun[i] * corr[i] + rs;
    }
#pragma unroll
    for (int i = 0; i < 8; ++i) {
      o0[i] *= corr[i]; o1[i] *= corr[i]; o2[i] *= corr[i]; o3[i] *= corr[i];
    }
    // C-layout -> A-layout transpose of P through wave-private LDS slice
#pragma unroll
    for (int i = 0; i < 8; ++i) {
      pbuf[(rowoff + i) * 32 + colb]      = f2bf(s0[i]);
      pbuf[(rowoff + i) * 32 + 16 + colb] = f2bf(s1[i]);
    }
    asm volatile("s_wait_dscnt 0x0" ::: "memory");
    v16bf pa = load_a(pbuf, 0, 32, 0, lane);
    // O(16x64) += P(16x32) x V(32x64) from LDS V tile
    o0 = wmma_bf16(pa, load_b(&sV[buf][0],  0, LDV, 0, lane), o0);
    o1 = wmma_bf16(pa, load_b(&sV[buf][0], 16, LDV, 0, lane), o1);
    o2 = wmma_bf16(pa, load_b(&sV[buf][0], 32, LDV, 0, lane), o2);
    o3 = wmma_bf16(pa, load_b(&sV[buf][0], 48, LDV, 0, lane), o3);
    __syncthreads();
    buf ^= 1;
  }
  v8f of[4] = {o0, o1, o2, o3};
#pragma unroll
  for (int t = 0; t < 4; ++t)
#pragma unroll
    for (int i = 0; i < 8; ++i) {
      int row = m0 + rowoff + i;
      int col = h * HEAD_D + t * 16 + colb;
      ob[(size_t)row * D_MODEL + col] = f2bf(of[t][i] / lrun[i]);
    }
}

// -------------------------------------------- output projection + residual
__global__ void fc_gemm(const __bf16* __restrict__ oin, const __bf16* __restrict__ wf,
                        const float* __restrict__ bias, const float* __restrict__ src,
                        float* __restrict__ yf) {
  __shared__ __align__(16) __bf16 sA[2][64 * LDA];
  __shared__ __align__(16) __bf16 sB[2][128 * LDB];
  const int tid  = threadIdx.x;
  const int lane = tid & 31;
  const int warp = tid >> 5;
  const int m0blk = (blockIdx.x >> 3) * 64;
  const int n0blk = (blockIdx.x & 7) * 128;
  const int mw = warp >> 2;
  const int nw = warp & 3;
  const int m0 = m0blk + mw * 32;
  const int n0 = n0blk + nw * 32;
  const int row4 = tid >> 2;
  const int ch   = tid & 3;

  auto stage = [&](int buf, int k0) {
    async_copy16(&sA[buf][row4 * LDA + ch * 8],
                 oin + (size_t)(m0blk + row4) * D_MODEL + k0 + ch * 8);
    async_copy16(&sB[buf][row4 * LDB + ch * 8],
                 wf + (size_t)(n0blk + row4) * D_MODEL + k0 + ch * 8);
    async_copy16(&sB[buf][(64 + row4) * LDB + ch * 8],
                 wf + (size_t)(n0blk + 64 + row4) * D_MODEL + k0 + ch * 8);
  };

  v8f c00 = {}, c01 = {}, c10 = {}, c11 = {};
  int buf = 0;
  stage(0, 0);
  for (int k0 = 0; k0 < D_MODEL; k0 += 32) {
    wait_async0();
    __syncthreads();
    if (k0 + 32 < D_MODEL) stage(buf ^ 1, k0 + 32);
    v16bf a0 = load_a(&sA[buf][0], mw * 32,      LDA, 0, lane);
    v16bf a1 = load_a(&sA[buf][0], mw * 32 + 16, LDA, 0, lane);
    v16bf b0 = load_b(&sB[buf][0], nw * 32,      LDB, 0, lane);
    v16bf b1 = load_b(&sB[buf][0], nw * 32 + 16, LDB, 0, lane);
    c00 = wmma_bf16(a0, b0, c00);
    c01 = wmma_bf16(a0, b1, c01);
    c10 = wmma_bf16(a1, b0, c10);
    c11 = wmma_bf16(a1, b1, c11);
    __syncthreads();
    buf ^= 1;
  }
  const int colb = lane & 15;
  const int rowoff = (lane >> 4) * 8;
  v8f cf[2][2] = {{c00, c01}, {c10, c11}};
#pragma unroll
  for (int mi = 0; mi < 2; ++mi)
#pragma unroll
    for (int ni = 0; ni < 2; ++ni)
#pragma unroll
      for (int i = 0; i < 8; ++i) {
        int row = m0 + mi * 16 + rowoff + i;
        int col = n0 + ni * 16 + colb;
        size_t idx = (size_t)row * D_MODEL + col;
        yf[idx] = cf[mi][ni][i] + bias[col] + src[idx];
      }
}

// -------------------------------------------------------------- layer norm
__global__ void layernorm_kernel(const float* __restrict__ yf, float* __restrict__ out) {
  const int row = blockIdx.x;
  const float* r = yf + (size_t)row * D_MODEL;
  float s = 0.0f, s2 = 0.0f;
  for (int j = threadIdx.x; j < D_MODEL; j += 256) {
    float v = r[j];
    s += v; s2 += v * v;
  }
#pragma unroll
  for (int off = 16; off > 0; off >>= 1) {
    s  += __shfl_xor(s,  off, 32);
    s2 += __shfl_xor(s2, off, 32);
  }
  __shared__ float as[8], as2[8];
  const int warp = threadIdx.x >> 5, lane = threadIdx.x & 31;
  if (lane == 0) { as[warp] = s; as2[warp] = s2; }
  __syncthreads();
  if (warp == 0) {
    s  = (lane < 8) ? as[lane]  : 0.0f;
    s2 = (lane < 8) ? as2[lane] : 0.0f;
#pragma unroll
    for (int off = 4; off > 0; off >>= 1) {
      s  += __shfl_xor(s,  off, 32);
      s2 += __shfl_xor(s2, off, 32);
    }
    if (lane == 0) { as[0] = s; as2[0] = s2; }
  }
  __syncthreads();
  const float mean = as[0] * (1.0f / D_MODEL);
  const float var  = as2[0] * (1.0f / D_MODEL) - mean * mean;
  const float inv  = rsqrtf(var + 1e-5f);
  float* o = out + (size_t)row * D_MODEL;
  for (int j = threadIdx.x; j < D_MODEL; j += 256) o[j] = (r[j] - mean) * inv;
}

// ------------------------------------------------------------------ launcher
extern "C" void kernel_launch(void* const* d_in, const int* in_sizes, int n_in,
                              void* d_out, int out_size, void* d_ws, size_t ws_size,
                              hipStream_t stream) {
  const float* src = (const float*)d_in[0];
  const float* Wq  = (const float*)d_in[1];
  const float* bq  = (const float*)d_in[2];
  const float* Wk  = (const float*)d_in[3];
  const float* bk  = (const float*)d_in[4];
  const float* Wv  = (const float*)d_in[5];
  const float* bv  = (const float*)d_in[6];
  const float* Wfc = (const float*)d_in[7];
  const float* bfc = (const float*)d_in[8];
  (void)in_sizes; (void)n_in; (void)out_size; (void)ws_size;

  // Workspace carve-up (~36 MiB total)
  __bf16* xb  = (__bf16*)d_ws;                 // 2048x1024
  __bf16* wqb = xb  + (size_t)N_TOK * D_MODEL;
  __bf16* wkb = wqb + (size_t)D_MODEL * D_MODEL;
  __bf16* wvb = wkb + (size_t)D_MODEL * D_MODEL;
  __bf16* wfb = wvb + (size_t)D_MODEL * D_MODEL;
  __bf16* qb  = wfb + (size_t)D_MODEL * D_MODEL; // 2048x1024
  __bf16* kbm = qb  + (size_t)N_TOK * D_MODEL;   // 2048x1024
  __bf16* vT  = kbm + (size_t)N_TOK * D_MODEL;   // 1024x2048 (transposed)
  __bf16* ob  = vT  + (size_t)N_TOK * D_MODEL;   // 2048x1024
  float*  yf  = (float*)(ob + (size_t)N_TOK * D_MODEL); // 2048x1024 f32

  cvt_f32_to_bf16<<<2048, 256, 0, stream>>>(src, xb,  N_TOK * D_MODEL);
  cvt_f32_to_bf16<<<1024, 256, 0, stream>>>(Wq,  wqb, D_MODEL * D_MODEL);
  cvt_f32_to_bf16<<<1024, 256, 0, stream>>>(Wk,  wkb, D_MODEL * D_MODEL);
  cvt_f32_to_bf16<<<1024, 256, 0, stream>>>(Wv,  wvb, D_MODEL * D_MODEL);
  cvt_f32_to_bf16<<<1024, 256, 0, stream>>>(Wfc, wfb, D_MODEL * D_MODEL);

  // 32 M-blocks x 8 N-blocks = 256 blocks; z = Q/K/V
  qkv_gemm<<<dim3(256, 1, 3), 256, 0, stream>>>(xb, wqb, wkb, wvb, bq, bk, bv,
                                                qb, kbm, vT);
  // 16 heads x 16 q-blocks (128 rows each) = 256 blocks
  attn_kernel<<<256, 256, 0, stream>>>(qb, kbm, vT, ob);
  fc_gemm<<<256, 256, 0, stream>>>(ob, wfb, bfc, src, yf);
  layernorm_kernel<<<N_TOK, 256, 0, stream>>>(yf, (float*)d_out);
}
